// TMFdecoder_17231408791843
// MI455X (gfx1250) — matrix operand order repaired and checked by reading, not compiled, not run on hardware
//
#include <hip/hip_runtime.h>
#include <hip/hip_bf16.h>

typedef __attribute__((ext_vector_type(16))) _Float16 v16h;
typedef __attribute__((ext_vector_type(8)))  _Float16 v8h;
typedef __attribute__((ext_vector_type(8)))  float    v8f;

#define CDIV(a,b) (((a)+(b)-1)/(b))
#define LEAKF 0.01f

__device__ __forceinline__ int imin(int a, int b) { return a < b ? a : b; }
__device__ __forceinline__ int imax(int a, int b) { return a > b ? a : b; }

__device__ __forceinline__ v16h mk_afrag(v8h a0, v8h a1) {
  v16h a;
#pragma unroll
  for (int j = 0; j < 8; ++j) { a[j] = a0[j]; a[8 + j] = a1[j]; }
  return a;
}

// ---------------------------------------------------------------------------
// WMMA GEMM workhorse. D = A * B (+ fused epilogue).
//   A: f16, [ntaps][Mpad][Kpad], row-major (weights, pre-packed, zero-padded)
//   B: f16, pixel-major activations: B[p*ldb + k]
//   ntaps==9 -> implicit-GEMM 3x3 conv, zero padding via a shared zero row
//   (branchless inner loop) + software-pipelined K loop (loads for k+32 in
//   flight behind the WMMAs for k).
// Each wave computes a 16 x (16*NT) f32 tile with v_wmma_f32_16x16x32_f16.
// ---------------------------------------------------------------------------
struct GemmP {
  const _Float16* A;
  const _Float16* B;
  const _Float16* zrow;  // >= Kpad zero halves
  int Mpad, Kpad, ldb;
  int ntaps, H, W, npix, Cout;
  const float* bias;     // [Cout]
  const float* bnsc;     // folded BN scale, or null
  const float* bnsh;     // folded BN shift, or null
  const float* embp;     // per-batch channel add [b][Cout], or null
  const float* mask;     // per-pixel multiplier [npix], or null
  int leaky;
  int smode;             // 0: f32 NCHW, 1: f16 pixel-major, 2: f16 NCHW
  float* of32;
  _Float16* of16;
  int ldo;
};

template <int NT>
__global__ void __launch_bounds__(128) k_gemm(GemmP P) {
  int wave = blockIdx.x * 4 + (threadIdx.x >> 5);
  int lane = threadIdx.x & 31;
  int mtiles = P.Mpad >> 4;
  int ntiles = CDIV(P.npix, 16 * NT);
  if (wave >= mtiles * ntiles) return;
  int mt = wave % mtiles;
  int nt = wave / mtiles;
  int M0 = mt << 4, N0 = nt * (16 * NT);
  int col = lane & 15, hi = lane >> 4;
  int HW = P.H * P.W;

  // Decompose each of my NT columns into (batch, y, x) once.
  int pb[NT], py[NT], px[NT];
  bool pv[NT];
#pragma unroll
  for (int t = 0; t < NT; ++t) {
    int p = N0 + t * 16 + col;
    pv[t] = p < P.npix;
    int pc = pv[t] ? p : 0;
    pb[t] = pc / HW;
    int rem = pc - pb[t] * HW;
    py[t] = rem / P.W;
    px[t] = rem - py[t] * P.W;
  }

  v8f acc[NT];
#pragma unroll
  for (int t = 0; t < NT; ++t)
#pragma unroll
    for (int j = 0; j < 8; ++j) acc[t][j] = 0.0f;

  const _Float16* Arow = P.A + (size_t)(M0 + col) * P.Kpad;
  size_t tapstride = (size_t)P.Mpad * P.Kpad;

  for (int tap = 0; tap < P.ntaps; ++tap) {
    int dy = (P.ntaps == 1) ? 0 : tap / 3 - 1;
    int dx = (P.ntaps == 1) ? 0 : tap % 3 - 1;
    // Branch-free per-tap B row pointers: halo/OOB columns -> shared zero row.
    const _Float16* bp[NT];
#pragma unroll
    for (int t = 0; t < NT; ++t) {
      int y = py[t] + dy, x = px[t] + dx;
      bool ok = pv[t] && y >= 0 && y < P.H && x >= 0 && x < P.W;
      bp[t] = ok ? P.B + (size_t)(pb[t] * HW + y * P.W + x) * P.ldb + hi * 16
                 : P.zrow + hi * 16;
    }
    const _Float16* At = Arow + tap * tapstride;

    // Software-pipelined K loop: preload k=0, then load k+32 behind WMMAs.
    // A fragment: lane L holds M = L&15, K = 16*(L>>4) + j (contiguous)
    // B fragment: lane L holds N = L&15, K = 16*(L>>4) + j (contiguous)
    v8h a0 = *(const v8h*)(At + hi * 16);
    v8h a1 = *(const v8h*)(At + hi * 16 + 8);
    v16h bb[NT];
#pragma unroll
    for (int t = 0; t < NT; ++t) bb[t] = *(const v16h*)(bp[t]);

    for (int k0 = 32; k0 < P.Kpad; k0 += 32) {
      v8h na0 = *(const v8h*)(At + k0 + hi * 16);
      v8h na1 = *(const v8h*)(At + k0 + hi * 16 + 8);
      v16h nb[NT];
#pragma unroll
      for (int t = 0; t < NT; ++t) nb[t] = *(const v16h*)(bp[t] + k0);
      v16h a = mk_afrag(a0, a1);
#pragma unroll
      for (int t = 0; t < NT; ++t)
        acc[t] = __builtin_amdgcn_wmma_f32_16x16x32_f16(
            false, a, false, bb[t], (short)0, acc[t], false, false);
      a0 = na0; a1 = na1;
#pragma unroll
      for (int t = 0; t < NT; ++t) bb[t] = nb[t];
    }
    {
      v16h a = mk_afrag(a0, a1);
#pragma unroll
      for (int t = 0; t < NT; ++t)
        acc[t] = __builtin_amdgcn_wmma_f32_16x16x32_f16(
            false, a, false, bb[t], (short)0, acc[t], false, false);
    }
  }

  // Epilogue. C/D layout: VGPR r holds M = M0 + 8*hi + r, N = lane&15.
  for (int t = 0; t < NT; ++t) {
    if (!pv[t]) continue;
    int p = N0 + t * 16 + col;
    int bb2 = pb[t], hw = py[t] * P.W + px[t];
    for (int r = 0; r < 8; ++r) {
      int m = M0 + hi * 8 + r;
      if (m >= P.Cout) break;
      float v = acc[t][r] + P.bias[m];
      if (P.embp) v += P.embp[bb2 * P.Cout + m];
      if (P.bnsc) v = v * P.bnsc[m] + P.bnsh[m];
      if (P.leaky) v = v >= 0.0f ? v : LEAKF * v;
      if (P.mask) v *= P.mask[p];
      if (P.smode == 0)
        P.of32[((size_t)bb2 * P.Cout + m) * HW + hw] = v;
      else if (P.smode == 1)
        P.of16[(size_t)p * P.ldo + m] = (_Float16)v;
      else
        P.of16[((size_t)bb2 * P.Cout + m) * HW + hw] = (_Float16)v;
    }
  }
}

// ---------------------------------------------------------------------------
// Support kernels
// ---------------------------------------------------------------------------

// conv0 last-two-channel sum, half-pixel bilinear 512 -> 32 (jax.image.resize)
__global__ void k_mask32(const float* conv0, float* out) {
  int i = blockIdx.x * blockDim.x + threadIdx.x;
  if (i >= 2 * 32 * 32) return;
  int bb = i >> 10, rem = i & 1023, y = rem >> 5, x = rem & 31;
  float py = (y + 0.5f) * 16.0f - 0.5f;
  float px = (x + 0.5f) * 16.0f - 0.5f;
  int y0 = (int)floorf(py); float ty = py - (float)y0;
  int x0 = (int)floorf(px); float tx = px - (float)x0;
  int y0c = imin(imax(y0, 0), 511), y1c = imin(imax(y0 + 1, 0), 511);
  int x0c = imin(imax(x0, 0), 511), x1c = imin(imax(x0 + 1, 0), 511);
  const float* c4 = conv0 + ((size_t)bb * 6 + 4) * 262144;
  const float* c5 = conv0 + ((size_t)bb * 6 + 5) * 262144;
  auto S = [&](int yy, int xx) { int o = yy * 512 + xx; return c4[o] + c5[o]; };
  float v0 = S(y0c, x0c) * (1.f - tx) + S(y0c, x1c) * tx;
  float v1 = S(y1c, x0c) * (1.f - tx) + S(y1c, x1c) * tx;
  out[i] = v0 * (1.f - ty) + v1 * ty;
}

// NCHW f32 -> pixel-major f16
__global__ void k_topc(const float* src, _Float16* dst, int C, int HW, int ldo,
                       long total) {
  long i = (long)blockIdx.x * blockDim.x + threadIdx.x;
  if (i >= total) return;
  int hw = (int)(i % HW);
  long t = i / HW;
  int c = (int)(t % C);
  int bb = (int)(t / C);
  dst[((size_t)bb * HW + hw) * ldo + c] = (_Float16)src[i];
}

// pack 1x1 weight (O x C f32) -> [Mpad][Kpad] f16 zero-padded
__global__ void k_pack1(const float* w, _Float16* out, int O, int C, int Mpad,
                        int Kpad) {
  long i = (long)blockIdx.x * blockDim.x + threadIdx.x;
  if (i >= (long)Mpad * Kpad) return;
  int k = (int)(i % Kpad), m = (int)(i / Kpad);
  out[i] = (m < O && k < C) ? (_Float16)w[(size_t)m * C + k] : (_Float16)0.0f;
}

// pack 3x3 weight (O,C,3,3 f32) -> [tap][Mpad][Kpad] f16 zero-padded
__global__ void k_pack3(const float* w, _Float16* out, int O, int C, int Mpad,
                        int Kpad) {
  long i = (long)blockIdx.x * blockDim.x + threadIdx.x;
  if (i >= 9L * Mpad * Kpad) return;
  long mk = (long)Mpad * Kpad;
  int tap = (int)(i / mk);
  long r = i % mk;
  int k = (int)(r % Kpad), m = (int)(r / Kpad);
  out[i] = (m < O && k < C) ? (_Float16)w[((size_t)m * C + k) * 9 + tap]
                            : (_Float16)0.0f;
}

// fold BN: scale = g*rsqrt(v+eps); shift = bt - m*scale
__global__ void k_bnf(const float* g, const float* v, const float* m,
                      const float* bt, float* sc, float* sh, int C) {
  int i = blockIdx.x * blockDim.x + threadIdx.x;
  if (i >= C) return;
  float s = g[i] * rsqrtf(v[i] + 1e-5f);
  sc[i] = s;
  sh[i] = bt[i] - m[i] * s;
}

__global__ void k_boxh(const float* in, float* out, int N, int H, int W, int rad) {
  long i = (long)blockIdx.x * blockDim.x + threadIdx.x;
  if (i >= (long)N * H * W) return;
  int x = (int)(i % W);
  long r = i / W;
  int y = (int)(r % H);
  long n = r / H;
  const float* row = in + ((size_t)n * H + y) * W;
  int lo = imax(x - rad, 0), hh = imin(x + rad, W - 1);
  float s = 0.f;
  for (int xx = lo; xx <= hh; ++xx) s += row[xx];
  out[i] = s;
}

__global__ void k_boxv(const float* in, float* out, int N, int H, int W, int rad) {
  long i = (long)blockIdx.x * blockDim.x + threadIdx.x;
  if (i >= (long)N * H * W) return;
  int x = (int)(i % W);
  long r = i / W;
  int y = (int)(r % H);
  long n = r / H;
  const float* img = in + (size_t)n * H * W;
  int lo = imax(y - rad, 0), hh = imin(y + rad, H - 1);
  float s = 0.f;
  for (int yy = lo; yy <= hh; ++yy) s += img[(size_t)yy * W + x];
  out[i] = s;
}

// tripool finalize: (sumF/cnt) / (sumM/cnt + 1e-6) -> XU channel slot (f16 PC)
__global__ void k_ppmfin(const float* featB, const float* maskB, _Float16* XU,
                         int rad, int slot) {
  int i = blockIdx.x * blockDim.x + threadIdx.x;
  if (i >= 2 * 256 * 1024) return;
  int hw = i & 1023, c = (i >> 10) & 255, bb = i >> 18;
  int y = hw >> 5, x = hw & 31;
  float cy = (float)(imin(y + rad, 31) - imax(y - rad, 0) + 1);
  float cx = (float)(imin(x + rad, 31) - imax(x - rad, 0) + 1);
  float cnt = cy * cx;
  float f = featB[((size_t)bb * 256 + c) * 1024 + hw] / cnt;
  float mm = maskB[bb * 1024 + hw] / cnt;
  XU[((size_t)bb * 1024 + hw) * 3072 + 2048 + slot * 256 + c] =
      (_Float16)(f / (mm + 1e-6f));
}

__global__ void k_mean(const float* x, float* emb, int C, int HW) {
  int i = blockIdx.x * blockDim.x + threadIdx.x;
  if (i >= 2 * C) return;
  const float* p = x + (size_t)i * HW;
  float s = 0.f;
  for (int j = 0; j < HW; ++j) s += p[j];
  emb[i] = s / (float)HW;
}

// embp[b][j] = emb_w[j,:] . emb[b,:] + emb_b[j]
__global__ void k_eproj(const float* emb, const float* w, const float* bias,
                        float* out, int r) {
  int i = blockIdx.x * blockDim.x + threadIdx.x;
  if (i >= 2 * r) return;
  int bb = i / r, j = i % r;
  const float* wr = w + (size_t)j * 256;
  const float* e = emb + bb * 256;
  float s = bias[j];
  for (int c = 0; c < 256; ++c) s += wr[c] * e[c];
  out[i] = s;
}

// align-corners bilinear resize, NCHW f32
__global__ void k_resize(const float* in, float* out, int C, int ih, int iw,
                         int oh, int ow) {
  long total = 2L * C * oh * ow;
  long i = (long)blockIdx.x * blockDim.x + threadIdx.x;
  if (i >= total) return;
  int ox = (int)(i % ow);
  long r = i / ow;
  int oy = (int)(r % oh);
  long t = r / oh;
  int c = (int)(t % C);
  int bb = (int)(t / C);
  float sy = (oh > 1) ? (float)(ih - 1) / (float)(oh - 1) : 0.f;
  float sx = (ow > 1) ? (float)(iw - 1) / (float)(ow - 1) : 0.f;
  float py = oy * sy, px = ox * sx;
  int y0 = imin((int)py, ih - 1), x0 = imin((int)px, iw - 1);
  int y1 = imin(y0 + 1, ih - 1), x1 = imin(x0 + 1, iw - 1);
  float ty = py - (float)y0, tx = px - (float)x0;
  const float* im = in + ((size_t)bb * C + c) * ih * iw;
  float v = im[(size_t)y0 * iw + x0] * (1.f - tx) * (1.f - ty) +
            im[(size_t)y0 * iw + x1] * tx * (1.f - ty) +
            im[(size_t)y1 * iw + x0] * (1.f - tx) * ty +
            im[(size_t)y1 * iw + x1] * tx * ty;
  out[i] = v;
}

// cat = concat(pixel_shuffle(high,2), low) -> pixel-major f16 (zero channel pad)
__global__ void k_cat(const float* high, const float* low, _Float16* cat,
                      int Chigh, int Clow, int H, int W, int ldc) {
  long total = 2L * H * W * ldc;
  long i = (long)blockIdx.x * blockDim.x + threadIdx.x;
  if (i >= total) return;
  int c = (int)(i % ldc);
  long p = i / ldc;
  int HW = H * W;
  int bb = (int)(p / HW);
  int rem = (int)(p - (long)bb * HW);
  int y = rem / W, x = rem % W;
  int C4 = Chigh >> 2;
  float v = 0.f;
  if (c < C4) {
    int hh = H >> 1, wh = W >> 1;
    int sc = c * 4 + (y & 1) * 2 + (x & 1);
    v = high[(((size_t)bb * Chigh + sc) * hh + (y >> 1)) * wh + (x >> 1)];
  } else if (c < C4 + Clow) {
    v = low[(((size_t)bb * Clow + (c - C4)) * H + y) * (size_t)W + x];
  }
  cat[i] = (_Float16)v;
}

// dynamic local filtering: out[b,gi*16+c,y,x] = sum_k w[b,gi*9+k,y,x]*cat[...shift]
__global__ void k_dyn(const _Float16* cr, const _Float16* wd, _Float16* dyn,
                      int inter, int G, int H, int W, int npix) {
  long total = (long)npix * inter;
  long i = (long)blockIdx.x * blockDim.x + threadIdx.x;
  if (i >= total) return;
  int ch = (int)(i % inter);
  long p = i / inter;
  int HW = H * W;
  int bb = (int)(p / HW);
  int rem = (int)(p - (long)bb * HW);
  int y = rem / W, x = rem % W;
  int gi = ch >> 4;  // inter/G == 16 always
  const _Float16* wbase = wd + ((size_t)bb * G * 9 + gi * 9) * (size_t)HW + rem;
  float acc = 0.f;
#pragma unroll
  for (int k = 0; k < 9; ++k) {
    int yy = y + k / 3 - 1, xx = x + k % 3 - 1;
    if (yy < 0 || yy >= H || xx < 0 || xx >= W) continue;
    float wv = (float)wbase[(size_t)k * HW];
    float cv = (float)cr[((size_t)bb * HW + (size_t)yy * W + xx) * inter + ch];
    acc += wv * cv;
  }
  dyn[(size_t)p * inter + ch] = (_Float16)acc;
}

__global__ void k_zero16(_Float16* p, long n) {
  long i = (long)blockIdx.x * blockDim.x + threadIdx.x;
  if (i < n) p[i] = (_Float16)0.0f;
}

// final 1x1 conv 16 -> 1
__global__ void k_final(const _Float16* y16, const float* w2, const float* b2,
                        float* out, int npix) {
  int i = blockIdx.x * blockDim.x + threadIdx.x;
  if (i >= npix) return;
  const _Float16* row = y16 + (size_t)i * 16;
  float s = b2[0];
#pragma unroll
  for (int o = 0; o < 16; ++o) s += w2[o] * (float)row[o];
  out[i] = s;
}

// ---------------------------------------------------------------------------
// Host orchestration
// ---------------------------------------------------------------------------

// Input indices: jax pytree flattening (dicts: sorted keys; lists: in order)
#define IN_CONV0 0
#define IN_CONV1 1
#define IN_CONV2 2
#define IN_CONV5 3
#define G2 4
#define G3 18
#define G4 32
#define GLF_BT 0
#define GLF_G 1
#define GLF_M 2
#define GLF_V 3
#define GLF_C1B 4
#define GLF_C1W 5
#define GLF_C2B 6
#define GLF_C2W 7
#define GLF_EMBB 8
#define GLF_EMBW 9
#define GLF_OUTB 10
#define GLF_OUTW 11
#define GLF_REDB 12
#define GLF_REDW 13
#define PPM0 46  // per ppm (stride 6): b, bn_bt, bn_g, bn_m, bn_v, w
#define UP1_B1 70
#define UP1_B2 71
#define UP1_BN1 72  // bt,g,m,v
#define UP1_BN2 76
#define UP1_W1 80
#define UP1_W2 81
#define UP4_B1 82
#define UP4_B2 83
#define UP4_W1 84
#define UP4_W2 85

static void gemm_launch(hipStream_t s, const _Float16* A, const _Float16* B,
                        const _Float16* zrow, int Mpad, int Kpad, int ldb,
                        int ntaps, int H, int W, int Cout, const float* bias,
                        const float* bnsc, const float* bnsh, const float* embp,
                        const float* mask, int leaky, int smode, float* of32,
                        _Float16* of16, int ldo) {
  GemmP P;
  P.A = A; P.B = B; P.zrow = zrow; P.Mpad = Mpad; P.Kpad = Kpad; P.ldb = ldb;
  P.ntaps = ntaps; P.H = H; P.W = W; P.npix = 2 * H * W; P.Cout = Cout;
  P.bias = bias; P.bnsc = bnsc; P.bnsh = bnsh; P.embp = embp; P.mask = mask;
  P.leaky = leaky; P.smode = smode; P.of32 = of32; P.of16 = of16; P.ldo = ldo;
  // Wider N-tiles when K is shallow (no double-buffer live-range pressure):
  // 16x128 tiles amortize A loads on the 512x512 stages; deep-K GEMMs keep
  // 16x64 so acc + double-buffered B frags stay within the register budget.
  if (Kpad <= 64) {
    int waves = (Mpad / 16) * CDIV(P.npix, 128);
    k_gemm<8><<<CDIV(waves, 4), 128, 0, s>>>(P);
  } else {
    int waves = (Mpad / 16) * CDIV(P.npix, 64);
    k_gemm<4><<<CDIV(waves, 4), 128, 0, s>>>(P);
  }
}

extern "C" void kernel_launch(void* const* d_in, const int* in_sizes, int n_in,
                              void* d_out, int out_size, void* d_ws,
                              size_t ws_size, hipStream_t stream) {
  (void)in_sizes; (void)n_in; (void)out_size; (void)ws_size;
  auto F = [&](int i) { return (const float*)d_in[i]; };
  const float* conv0 = F(IN_CONV0);
  const float* conv1 = F(IN_CONV1);
  const float* conv2 = F(IN_CONV2);
  const float* conv5 = F(IN_CONV5);

  char* base = (char*)d_ws;
  size_t off = 0;
  auto alloc = [&](size_t bytes) -> char* {
    char* p = base + off;
    off += (bytes + 255) & ~(size_t)255;
    return p;
  };
  auto afp = [&](size_t n) -> float* { return (float*)alloc(n * 4); };
  auto ahp = [&](size_t n) -> _Float16* { return (_Float16*)alloc(n * 2); };

  // ---- folded BN params
  float *ppm_sc[4], *ppm_sh[4];
  for (int i = 0; i < 4; ++i) { ppm_sc[i] = afp(256); ppm_sh[i] = afp(256); }
  float* u1sc1 = afp(256); float* u1sh1 = afp(256);
  float* u1sc2 = afp(256); float* u1sh2 = afp(256);
  float* g2sc = afp(256);  float* g2sh = afp(256);
  float* g3sc = afp(64);   float* g3sh = afp(64);
  float* g4sc = afp(32);   float* g4sh = afp(32);

  // ---- packed f16 weights
  _Float16* zrow = ahp(4096);  // shared zero row for halo columns
  _Float16* ppmw[4];
  for (int i = 0; i < 4; ++i) ppmw[i] = ahp((size_t)256 * 2048);
  _Float16* u1w1p = ahp(9L * 256 * 3072);
  _Float16* u1w2p = ahp(9L * 256 * 256);
  _Float16* g2red = ahp((size_t)256 * 320);
  _Float16* g2c1 = ahp((size_t)64 * 256);
  _Float16* g2c2p = ahp(9L * 144 * 64);
  _Float16* g2out = ahp((size_t)256 * 256);
  _Float16* g3red = ahp((size_t)256 * 192);
  _Float16* g3c1 = ahp((size_t)64 * 256);
  _Float16* g3c2p = ahp(9L * 144 * 64);
  _Float16* g3out = ahp((size_t)64 * 256);
  _Float16* g4red = ahp((size_t)32 * 32);
  _Float16* g4c1 = ahp((size_t)16 * 32);
  _Float16* g4c2p = ahp(9L * 32 * 32);
  _Float16* g4out = ahp((size_t)32 * 32);
  _Float16* u4w1p = ahp(9L * 16 * 32);

  // ---- activations / scratch
  float* mask32 = afp(2048);
  _Float16* XU = ahp((size_t)2048 * 3072);  // [pix][3072] f16
  float* featM = afp((size_t)2 * 256 * 1024);
  float* btmp = afp((size_t)2 * 256 * 1024);
  float* featB = afp((size_t)2 * 256 * 1024);
  float* mtmp = afp(2048);
  float* maskB = afp(2048);
  _Float16* Xmid = ahp((size_t)2048 * 256);
  float* xup1 = afp((size_t)2 * 256 * 1024);
  float* emb = afp(512);
  float* e2 = afp(128);
  float* e3 = afp(128);
  float* e4 = afp(16);
  float* xhi2 = afp((size_t)2 * 256 * 4096);
  _Float16* SL_CAT = ahp(25165824);  // max cat / p4 slab
  _Float16* SL_CR = ahp(33554432);   // max cat-reduced slab
  _Float16* SL_T = ahp(16777216);    // max t / y16 slab
  _Float16* SL_WD = ahp(18874368);   // max dynamic-weight slab (f16 NCHW)
  _Float16* SL_DYN = ahp(33554432);  // max dyn-filter output slab
  float* OUT_A = afp(16777216);      // glf out ping
  float* OUT_B = afp(16777216);      // glf out pong

#define EL(n, k, ...) k<<<CDIV((long)(n), 256), 256, 0, stream>>>(__VA_ARGS__)

  EL(4096, k_zero16, zrow, 4096L);

  // ---- BN folds
  for (int i = 0; i < 4; ++i)
    EL(256, k_bnf, F(PPM0 + 6 * i + 2), F(PPM0 + 6 * i + 4), F(PPM0 + 6 * i + 3),
       F(PPM0 + 6 * i + 1), ppm_sc[i], ppm_sh[i], 256);
  EL(256, k_bnf, F(UP1_BN1 + 1), F(UP1_BN1 + 3), F(UP1_BN1 + 2), F(UP1_BN1 + 0),
     u1sc1, u1sh1, 256);
  EL(256, k_bnf, F(UP1_BN2 + 1), F(UP1_BN2 + 3), F(UP1_BN2 + 2), F(UP1_BN2 + 0),
     u1sc2, u1sh2, 256);
  EL(256, k_bnf, F(G2 + GLF_G), F(G2 + GLF_V), F(G2 + GLF_M), F(G2 + GLF_BT),
     g2sc, g2sh, 256);
  EL(64, k_bnf, F(G3 + GLF_G), F(G3 + GLF_V), F(G3 + GLF_M), F(G3 + GLF_BT),
     g3sc, g3sh, 64);
  EL(32, k_bnf, F(G4 + GLF_G), F(G4 + GLF_V), F(G4 + GLF_M), F(G4 + GLF_BT),
     g4sc, g4sh, 32);

  // ---- weight packs
  for (int i = 0; i < 4; ++i)
    EL((size_t)256 * 2048, k_pack1, F(PPM0 + 6 * i + 5), ppmw[i], 256, 2048, 256, 2048);
  EL(9L * 256 * 3072, k_pack3, F(UP1_W1), u1w1p, 256, 3072, 256, 3072);
  EL(9L * 256 * 256, k_pack3, F(UP1_W2), u1w2p, 256, 256, 256, 256);
  EL((size_t)256 * 320, k_pack1, F(G2 + GLF_REDW), g2red, 256, 320, 256, 320);
  EL((size_t)64 * 256, k_pack1, F(G2 + GLF_C1W), g2c1, 64, 256, 64, 256);
  EL(9L * 144 * 64, k_pack3, F(G2 + GLF_C2W), g2c2p, 144, 64, 144, 64);
  EL((size_t)256 * 256, k_pack1, F(G2 + GLF_OUTW), g2out, 256, 256, 256, 256);
  EL((size_t)256 * 192, k_pack1, F(G3 + GLF_REDW), g3red, 256, 192, 256, 192);
  EL((size_t)64 * 256, k_pack1, F(G3 + GLF_C1W), g3c1, 64, 256, 64, 256);
  EL(9L * 144 * 64, k_pack3, F(G3 + GLF_C2W), g3c2p, 144, 64, 144, 64);
  EL((size_t)64 * 256, k_pack1, F(G3 + GLF_OUTW), g3out, 64, 256, 64, 256);
  EL((size_t)32 * 32, k_pack1, F(G4 + GLF_REDW), g4red, 32, 22, 32, 32);
  EL((size_t)16 * 32, k_pack1, F(G4 + GLF_C1W), g4c1, 8, 32, 16, 32);
  EL(9L * 32 * 32, k_pack3, F(G4 + GLF_C2W), g4c2p, 18, 8, 32, 32);
  EL((size_t)32 * 32, k_pack1, F(G4 + GLF_OUTW), g4out, 32, 32, 32, 32);
  EL(9L * 16 * 32, k_pack3, F(UP4_W1), u4w1p, 16, 32, 16, 32);

  // ---- stage 1: mask + conv5 to pixel-major f16
  EL(2048, k_mask32, conv0, mask32);
  EL(2L * 2048 * 1024, k_topc, conv5, XU, 2048, 1024, 3072, 2L * 2048 * 1024);

  // ---- stage 2: PPM (tripool at k = 31,17,11,5)
  static const int kk[4] = {31, 17, 11, 5};
  for (int i = 0; i < 4; ++i) {
    gemm_launch(stream, ppmw[i], XU, zrow, 256, 2048, 3072, 1, 32, 32, 256,
                F(PPM0 + 6 * i), ppm_sc[i], ppm_sh[i], nullptr, mask32, 1, 0,
                featM, nullptr, 0);
    int rad = (kk[i] - 1) / 2;
    EL((size_t)512 * 1024, k_boxh, featM, btmp, 512, 32, 32, rad);
    EL((size_t)512 * 1024, k_boxv, btmp, featB, 512, 32, 32, rad);
    EL(2048, k_boxh, mask32, mtmp, 2, 32, 32, rad);
    EL(2048, k_boxv, mtmp, maskB, 2, 32, 32, rad);
    EL((size_t)2 * 256 * 1024, k_ppmfin, featB, maskB, XU, rad, i);
  }

  // ---- stage 3: up1 (two 3x3 convs with BN + leaky)
  gemm_launch(stream, u1w1p, XU, zrow, 256, 3072, 3072, 9, 32, 32, 256,
              F(UP1_B1), u1sc1, u1sh1, nullptr, nullptr, 1, 1, nullptr, Xmid,
              256);
  gemm_launch(stream, u1w2p, Xmid, zrow, 256, 256, 256, 9, 32, 32, 256,
              F(UP1_B2), u1sc2, u1sh2, nullptr, nullptr, 1, 0, xup1, nullptr,
              0);

  // ---- stage 4: embedding + projections + align-corners upsample
  EL(512, k_mean, xup1, emb, 256, 1024);
  EL(128, k_eproj, emb, F(G2 + GLF_EMBW), F(G2 + GLF_EMBB), e2, 64);
  EL(128, k_eproj, emb, F(G3 + GLF_EMBW), F(G3 + GLF_EMBB), e3, 64);
  EL(16, k_eproj, emb, F(G4 + GLF_EMBW), F(G4 + GLF_EMBB), e4, 8);
  EL(2L * 256 * 4096, k_resize, xup1, xhi2, 256, 32, 32, 64, 64);

  // ---- stage 5: glf2 (high = xhi2 256@64, low = conv2 256@128) -> OUT_A
  EL(2L * 128 * 128 * 320, k_cat, xhi2, conv2, SL_CAT, 256, 256, 128, 128, 320);
  gemm_launch(stream, g2red, SL_CAT, zrow, 256, 320, 320, 1, 128, 128, 256,
              F(G2 + GLF_REDB), nullptr, nullptr, nullptr, nullptr, 0, 1,
              nullptr, SL_CR, 256);
  gemm_launch(stream, g2c1, SL_CR, zrow, 64, 256, 256, 1, 128, 128, 64,
              F(G2 + GLF_C1B), nullptr, nullptr, e2, nullptr, 1, 1, nullptr,
              SL_T, 64);
  gemm_launch(stream, g2c2p, SL_T, zrow, 144, 64, 64, 9, 128, 128, 144,
              F(G2 + GLF_C2B), nullptr, nullptr, nullptr, nullptr, 0, 2,
              nullptr, SL_WD, 0);
  EL(32768L * 256, k_dyn, SL_CR, SL_WD, SL_DYN, 256, 16, 128, 128, 32768);
  gemm_launch(stream, g2out, SL_DYN, zrow, 256, 256, 256, 1, 128, 128, 256,
              F(G2 + GLF_OUTB), g2sc, g2sh, nullptr, nullptr, 1, 0, OUT_A,
              nullptr, 0);

  // ---- stage 6: glf3 (high = OUT_A 256@128, low = conv1 128@256) -> OUT_B
  EL(2L * 256 * 256 * 192, k_cat, OUT_A, conv1, SL_CAT, 256, 128, 256, 256, 192);
  gemm_launch(stream, g3red, SL_CAT, zrow, 256, 192, 192, 1, 256, 256, 256,
              F(G3 + GLF_REDB), nullptr, nullptr, nullptr, nullptr, 0, 1,
              nullptr, SL_CR, 256);
  gemm_launch(stream, g3c1, SL_CR, zrow, 64, 256, 256, 1, 256, 256, 64,
              F(G3 + GLF_C1B), nullptr, nullptr, e3, nullptr, 1, 1, nullptr,
              SL_T, 64);
  gemm_launch(stream, g3c2p, SL_T, zrow, 144, 64, 64, 9, 256, 256, 144,
              F(G3 + GLF_C2B), nullptr, nullptr, nullptr, nullptr, 0, 2,
              nullptr, SL_WD, 0);
  EL(131072L * 256, k_dyn, SL_CR, SL_WD, SL_DYN, 256, 16, 256, 256, 131072);
  gemm_launch(stream, g3out, SL_DYN, zrow, 64, 256, 256, 1, 256, 256, 64,
              F(G3 + GLF_OUTB), g3sc, g3sh, nullptr, nullptr, 1, 0, OUT_B,
              nullptr, 0);

  // ---- stage 7: glf4 (high = OUT_B 64@256, low = conv0 6@512) -> OUT_A
  EL(2L * 512 * 512 * 32, k_cat, OUT_B, conv0, SL_CAT, 64, 6, 512, 512, 32);
  gemm_launch(stream, g4red, SL_CAT, zrow, 32, 32, 32, 1, 512, 512, 32,
              F(G4 + GLF_REDB), nullptr, nullptr, nullptr, nullptr, 0, 1,
              nullptr, SL_CR, 32);
  EL(524288L * 32, k_zero16, SL_T, 524288L * 32);  // pad cols of t must be 0
  gemm_launch(stream, g4c1, SL_CR, zrow, 16, 32, 32, 1, 512, 512, 8,
              F(G4 + GLF_C1B), nullptr, nullptr, e4, nullptr, 1, 1, nullptr,
              SL_T, 32);
  gemm_launch(stream, g4c2p, SL_T, zrow, 32, 32, 32, 9, 512, 512, 18,
              F(G4 + GLF_C2B), nullptr, nullptr, nullptr, nullptr, 0, 2,
              nullptr, SL_WD, 0);
  EL(524288L * 32, k_dyn, SL_CR, SL_WD, SL_DYN, 32, 2, 512, 512, 524288);
  gemm_launch(stream, g4out, SL_DYN, zrow, 32, 32, 32, 1, 512, 512, 32,
              F(G4 + GLF_OUTB), g4sc, g4sh, nullptr, nullptr, 1, 0, OUT_A,
              nullptr, 0);

  // ---- stage 8: up4 (3x3 conv 32->16 + leaky, then 1x1 16->1)
  EL(2L * 32 * 262144, k_topc, OUT_A, SL_CAT, 32, 262144, 32, 2L * 32 * 262144);
  gemm_launch(stream, u4w1p, SL_CAT, zrow, 16, 32, 32, 9, 512, 512, 16,
              F(UP4_B1), nullptr, nullptr, nullptr, nullptr, 1, 1, nullptr,
              SL_T, 16);
  EL(524288, k_final, SL_T, F(UP4_W2), F(UP4_B2), (float*)d_out, 524288);
#undef EL
}
